// EnhancedExternalMemoryBank_39908836115152
// MI455X (gfx1250) — compile-verified, matrix operand order checked by reading, not run on hardware
//
#include <hip/hip_runtime.h>
#include <hip/hip_bf16.h>
#include <stdint.h>

// ---------------------------------------------------------------------------
// EnhancedExternalMemoryBank retrieve() for MI455X (gfx1250, wave32, WMMA)
//   H=8 heads, S=16384 chunks, C=8, Dh=64, N=4096 queries, k=32 (4 chunks)
// Pipeline:
//   K1a: queries f32 -> f16 (natural [n][h*64+d] layout)
//   K1b: chunk mean over C + f32->f16  (ck_h[h][s][d])
//   K2 : per-head GEMM (v_wmma_f32_16x16x32_f16, K=64 via 2 k-steps) fused
//        with streaming per-row top-4; B stream double-buffered in registers
//        (2x unrolled, alternating buffers) to hide L2 latency at low occupancy
//   K3 : gather 4 chunks of keys+vals per (n,h) -> [2, N*H, 32, 64]
// ---------------------------------------------------------------------------

typedef __attribute__((ext_vector_type(16))) _Float16 v16h;
typedef __attribute__((ext_vector_type(8)))  _Float16 v8h;
typedef __attribute__((ext_vector_type(4)))  _Float16 v4h;
typedef __attribute__((ext_vector_type(8)))  float    v8f;

#define HEADS 8
#define SSIZE 16384
#define CSIZE 8
#define DH    64
#define NQ    4096
#define DM    512
#define MT    2          // row tiles (of 16) per wave
#define NSB   (SSIZE / 16)

__device__ __forceinline__ uint32_t umax_(uint32_t a, uint32_t b) { return a > b ? a : b; }
__device__ __forceinline__ uint32_t umin_(uint32_t a, uint32_t b) { return a < b ? a : b; }

// Branchless insert of p into descending sorted top-4 {t0>=t1>=t2>=t3}.
// Lowers to fused v_minmax_u32 chains; identity when p <= t3.
__device__ __forceinline__ void ins4(uint32_t p, uint32_t& t0, uint32_t& t1,
                                     uint32_t& t2, uint32_t& t3) {
  uint32_t n0 = umax_(t0, p);
  uint32_t n1 = umax_(t1, umin_(t0, p));
  uint32_t n2 = umax_(t2, umin_(t1, p));
  uint32_t n3 = umax_(t3, umin_(t2, p));
  t0 = n0; t1 = n1; t2 = n2; t3 = n3;
}

// Per-lane share of a 16x32 f16 WMMA operand from a row-major [row][64] array:
// two contiguous 16B loads at half-offsets c0 and c0+16.
__device__ __forceinline__ v16h ldpair(const _Float16* __restrict__ p, int c0) {
  v8h lo = *(const v8h*)(p + c0);
  v8h hi = *(const v8h*)(p + c0 + 16);
  v16h r;
#pragma unroll
  for (int i = 0; i < 8; ++i) { r[i] = lo[i]; r[i + 8] = hi[i]; }
  return r;
}

// One 16-column tile: 2*MT WMMAs (independent accumulators back-to-back so
// later WMMAs fill earlier ones' WMMA->VALU hazard windows), then pack +
// branchless top-4 update.
__device__ __forceinline__ void process_tile(const v16h* a0, const v16h* a1,
                                             v16h b0, v16h b1, uint32_t sidx,
                                             uint32_t (*tk)[8][4]) {
  v8f acc[MT];
#pragma unroll
  for (int t = 0; t < MT; ++t) {
    v8f z = {};
    z = __builtin_amdgcn_wmma_f32_16x16x32_f16(
        false, a0[t], false, b0, (short)0, z, false, false);
    acc[t] = __builtin_amdgcn_wmma_f32_16x16x32_f16(
        false, a1[t], false, b1, (short)0, z, false, false);
  }
#pragma unroll
  for (int t = 0; t < MT; ++t) {
#pragma unroll
    for (int r = 0; r < 8; ++r) {
      int u = __float_as_int(acc[t][r]);
      // monotone uint key: positives set sign bit, negatives flip all bits
      uint32_t key = (uint32_t)u ^ (uint32_t)((u >> 31) | 0x80000000);
      uint32_t p = (key & 0xFFFFC000u) | sidx;   // 18b key | 14b column
      ins4(p, tk[t][r][0], tk[t][r][1], tk[t][r][2], tk[t][r][3]);
    }
  }
}

// ---- Kernel 1a: queries f32 -> f16 ----------------------------------------
__global__ void __launch_bounds__(256)
qcvt_kernel(const float* __restrict__ q, _Float16* __restrict__ qh) {
  int i = blockIdx.x * 256 + threadIdx.x;           // float4 index
  float4 v = ((const float4*)q)[i];
  v4h o;
  o[0] = (_Float16)v.x; o[1] = (_Float16)v.y;
  o[2] = (_Float16)v.z; o[3] = (_Float16)v.w;
  *(v4h*)(qh + (size_t)i * 4) = o;
}

// ---- Kernel 1b: chunk mean over C, f32 -> f16 -----------------------------
__global__ void __launch_bounds__(256)
chunk_mean_kernel(const float* __restrict__ mk, _Float16* __restrict__ ckh) {
  size_t gid = (size_t)blockIdx.x * 256 + threadIdx.x;  // over H*S*DH
  int d = (int)(gid & 63);
  size_t hs = gid >> 6;                                  // h*S + s
  const float* p = mk + hs * (size_t)(CSIZE * DH) + d;
  float acc = 0.f;
#pragma unroll
  for (int c = 0; c < CSIZE; ++c) acc += p[(size_t)c * DH];
  ckh[gid] = (_Float16)(acc * 0.125f);
}

// ---- Kernel 2: fused WMMA score GEMM + streaming top-4 --------------------
__global__ void __launch_bounds__(32)
score_topk_kernel(const _Float16* __restrict__ qh,
                  const _Float16* __restrict__ ckh,
                  uint32_t* __restrict__ idx_out) {
  const int lane    = threadIdx.x;        // one wave32 per workgroup
  const int h       = blockIdx.y;
  const int rowbase = blockIdx.x * (16 * MT);
  const int halfw   = lane >> 4;          // which 16-lane half
  const int lr      = lane & 15;
  const int c0      = halfw * 8;          // K-offset pattern for this half

  // A tiles: 16 rows x K=64 each, per-lane = 4x b128 loads
  v16h a0[MT], a1[MT];
#pragma unroll
  for (int t = 0; t < MT; ++t) {
    const _Float16* qp = qh + (size_t)(rowbase + t * 16 + lr) * DM + h * DH;
    a0[t] = ldpair(qp, c0);        // K 0..31 share
    a1[t] = ldpair(qp, c0 + 32);   // K 32..63 share
  }

  // per-lane top-4 packed (monotone-key[31:14] | column[13:0]) per row
  uint32_t tk[MT][8][4];
#pragma unroll
  for (int t = 0; t < MT; ++t)
#pragma unroll
    for (int r = 0; r < 8; ++r)
#pragma unroll
      for (int e = 0; e < 4; ++e) tk[t][r][e] = 0u;

  const _Float16* ckbase = ckh + ((size_t)h * SSIZE + lr) * DH;

  // Software-pipelined B stream: double-buffered registers, 2x unroll with
  // alternating buffers (no copy-backs). Loads for tile i+1 issue before the
  // compute on tile i, so s_wait_loadcnt covers a full compute phase.
  v16h b0a = ldpair(ckbase, c0);
  v16h b1a = ldpair(ckbase, c0 + 32);
  v16h b0b, b1b;

  for (int sb = 0; sb < NSB; sb += 2) {
    // prefetch tile sb+1 into buffer B (sb+1 <= NSB-1 always)
    {
      const _Float16* bp = ckbase + (size_t)(sb + 1) * (16 * DH);
      b0b = ldpair(bp, c0);
      b1b = ldpair(bp, c0 + 32);
    }
    process_tile(a0, a1, b0a, b1a, (uint32_t)(sb * 16 + lr), tk);

    // prefetch tile sb+2 into buffer A (clamped; final prefetch is dead)
    {
      int nx = (sb + 2 < NSB) ? (sb + 2) : (NSB - 1);
      const _Float16* bp = ckbase + (size_t)nx * (16 * DH);
      b0a = ldpair(bp, c0);
      b1a = ldpair(bp, c0 + 32);
    }
    process_tile(a0, a1, b0b, b1b, (uint32_t)((sb + 1) * 16 + lr), tk);
  }

  // merge top-4 lists across the 16 lanes of each half (log2(16)=4 steps)
#pragma unroll
  for (int off = 1; off < 16; off <<= 1) {
#pragma unroll
    for (int t = 0; t < MT; ++t)
#pragma unroll
      for (int r = 0; r < 8; ++r) {
        uint32_t p0 = __shfl_xor(tk[t][r][0], off, 32);
        uint32_t p1 = __shfl_xor(tk[t][r][1], off, 32);
        uint32_t p2 = __shfl_xor(tk[t][r][2], off, 32);
        uint32_t p3 = __shfl_xor(tk[t][r][3], off, 32);
        ins4(p0, tk[t][r][0], tk[t][r][1], tk[t][r][2], tk[t][r][3]);
        ins4(p1, tk[t][r][0], tk[t][r][1], tk[t][r][2], tk[t][r][3]);
        ins4(p2, tk[t][r][0], tk[t][r][1], tk[t][r][2], tk[t][r][3]);
        ins4(p3, tk[t][r][0], tk[t][r][1], tk[t][r][2], tk[t][r][3]);
      }
  }

  if (lr == 0) {
#pragma unroll
    for (int t = 0; t < MT; ++t)
#pragma unroll
      for (int r = 0; r < 8; ++r) {
        int n = rowbase + t * 16 + r + halfw * 8;  // D-matrix row mapping
        uint32_t* dst = idx_out + ((size_t)h * NQ + n) * 4;
        dst[0] = tk[t][r][0] & 0x3FFFu;
        dst[1] = tk[t][r][1] & 0x3FFFu;
        dst[2] = tk[t][r][2] & 0x3FFFu;
        dst[3] = tk[t][r][3] & 0x3FFFu;
      }
  }
}

// ---- Kernel 3: gather chosen chunks into [2, N*H, 32, 64] -----------------
__global__ void __launch_bounds__(256)
gather_kernel(const float* __restrict__ mk, const float* __restrict__ mv,
              const uint32_t* __restrict__ idx, float* __restrict__ out) {
  const int row = blockIdx.x;       // n*H + h (output row order)
  const int n = row / HEADS;
  const int h = row % HEADS;
  const uint32_t* ip = idx + ((size_t)h * NQ + n) * 4;
  const size_t NH = (size_t)NQ * HEADS;

  for (int i = threadIdx.x; i < 512; i += 256) {  // 512 float4 per part
    int kk = i >> 4;          // 0..31
    int d4 = i & 15;          // float4 within Dh
    int j  = kk >> 3;         // which retrieved chunk
    int c  = kk & 7;          // element within chunk
    uint32_t sid = ip[j];
    size_t srcf4 = (((size_t)h * SSIZE + sid) * CSIZE + c) * (DH / 4) + d4;
    size_t dstf4 = (size_t)row * 512 + (size_t)kk * (DH / 4) + d4;
    ((float4*)out)[dstf4]            = ((const float4*)mk)[srcf4];  // keys
    ((float4*)out)[dstf4 + NH * 512] = ((const float4*)mv)[srcf4];  // vals
  }
}

// ---------------------------------------------------------------------------
extern "C" void kernel_launch(void* const* d_in, const int* in_sizes, int n_in,
                              void* d_out, int out_size, void* d_ws, size_t ws_size,
                              hipStream_t stream) {
  const float* queries = (const float*)d_in[0];   // [2048,2,512]
  const float* mk      = (const float*)d_in[1];   // [8,16384,8,64]
  const float* mv      = (const float*)d_in[2];   // [8,16384,8,64]
  float* out = (float*)d_out;                     // [2, 32768, 32, 64]

  char* ws = (char*)d_ws;
  _Float16* qh   = (_Float16*)ws;                                  //  4 MB
  _Float16* ckh  = (_Float16*)(ws + (size_t)4  * 1024 * 1024);     // 16 MB
  uint32_t* idxb = (uint32_t*)(ws + (size_t)20 * 1024 * 1024);     // 0.5 MB

  // K1a: 4096*512 / 4 float4 = 524288 -> 2048 blocks
  qcvt_kernel<<<2048, 256, 0, stream>>>(queries, qh);
  // K1b: H*S*DH = 8388608 threads -> 32768 blocks
  chunk_mean_kernel<<<32768, 256, 0, stream>>>(mk, ckh);
  // K2: one wave per (32-row tile, head)
  score_topk_kernel<<<dim3(NQ / (16 * MT), HEADS), 32, 0, stream>>>(qh, ckh, idxb);
  // K3: one block per output row
  gather_kernel<<<NQ * HEADS, 256, 0, stream>>>(mk, mv, idxb, out);
}
